// CrossAttentionLayer_13975823582159
// MI455X (gfx1250) — matrix operand order repaired and checked by reading, not compiled, not run on hardware
//
#include <hip/hip_runtime.h>
#include <hip/hip_bf16.h>

// ---------------------------------------------------------------------------
// CDNA5 (gfx1250, wave32) cross-attention layer.
// GEMMs: v_wmma_f32_16x16x32_bf16 with async global->LDS B-panel staging.
// ---------------------------------------------------------------------------

typedef __attribute__((ext_vector_type(16))) __bf16       v16bf;
typedef __attribute__((ext_vector_type(8)))  float        v8f;
typedef __attribute__((ext_vector_type(8)))  unsigned int v8u;
typedef __attribute__((ext_vector_type(4)))  unsigned int v4u;
typedef __attribute__((ext_vector_type(4)))  int          v4i;

#define D_DIM 2048
#define M_DIM 1024
#define N_DIM 1024

// Async global->LDS (ASYNCcnt-tracked) if the toolchain exposes the builtins.
#if defined(__has_builtin)
#if __has_builtin(__builtin_amdgcn_global_load_async_to_lds_b128) && \
    __has_builtin(__builtin_amdgcn_s_wait_asynccnt)
#define CA_HAVE_ASYNC_LDS 1
#endif
#endif
#ifndef CA_HAVE_ASYNC_LDS
#define CA_HAVE_ASYNC_LDS 0
#endif

// float -> bf16 bits, round-to-nearest-even
static __device__ __forceinline__ unsigned short f2bf(float x) {
  unsigned u = __builtin_bit_cast(unsigned, x);
  unsigned r = u + 0x7FFFu + ((u >> 16) & 1u);
  return (unsigned short)(r >> 16);
}

// ---------------------------------------------------------------------------
// Prep: bias dots (u_p@w1, u_c@w2) + bf16 operand copies (+ K-major transposes)
// grid: (1024, 2), block: 256
// ---------------------------------------------------------------------------
__global__ __launch_bounds__(256)
void ca_prep_kernel(const float* __restrict__ u_p, const float* __restrict__ u_c,
                    const float* __restrict__ w_a,
                    unsigned short* __restrict__ Abf,   // (M x D)  bf16(u_p * w3)
                    unsigned short* __restrict__ Cbf,   // (N x D)  bf16(u_c)
                    unsigned short* __restrict__ CbfT,  // (D x N)  bf16(u_c)^T
                    unsigned short* __restrict__ PbfT,  // (D x M)  bf16(u_p)^T
                    float* __restrict__ rp, float* __restrict__ rc) {
  __shared__ float red[256];
  const int row = blockIdx.x;
  const int which = blockIdx.y;
  const int tid = threadIdx.x;
  const float* __restrict__ w3 = w_a + 2 * D_DIM;
  float s = 0.f;
  if (which == 0) {
    const float* __restrict__ w1 = w_a;
    for (int k = tid; k < D_DIM; k += 256) {
      float x = u_p[(size_t)row * D_DIM + k];
      s += x * w1[k];
      Abf[(size_t)row * D_DIM + k] = f2bf(x * w3[k]);
      PbfT[(size_t)k * M_DIM + row] = f2bf(x);
    }
  } else {
    const float* __restrict__ w2 = w_a + D_DIM;
    for (int k = tid; k < D_DIM; k += 256) {
      float x = u_c[(size_t)row * D_DIM + k];
      s += x * w2[k];
      Cbf[(size_t)row * D_DIM + k] = f2bf(x);
      CbfT[(size_t)k * N_DIM + row] = f2bf(x);
    }
  }
  red[tid] = s;
  __syncthreads();
  for (int off = 128; off > 0; off >>= 1) {
    if (tid < off) red[tid] += red[tid + off];
    __syncthreads();
  }
  if (tid == 0) (which ? rc : rp)[row] = red[0];
}

// ---------------------------------------------------------------------------
// bf16 WMMA GEMM: C[M x N] = A[M x K] * B[K x N] (+ rowBias + colBias)
// A row-major (lda = K-stride). B supplied N-major: Bt[n * ldb + k].
// Block = 4 waves = 64x64 output tile; wave w owns rows [64*br + 16*w, +16).
// B panel (64 N x 32 K bf16 = 4KB) is double-buffered in LDS, staged with
// async global->LDS copies (ASYNCcnt) and consumed via ds_load_b128.
// K must be a multiple of 32.
// ---------------------------------------------------------------------------
__global__ __launch_bounds__(128)
void ca_gemm_bf16_kernel(const unsigned short* __restrict__ A, int lda,
                         const unsigned short* __restrict__ Bt, int ldb,
                         float* __restrict__ C, int ldc,
                         int tilesN64, int K,
                         const float* __restrict__ biasRow,
                         const float* __restrict__ biasCol) {
  __shared__ __align__(16) unsigned short Bs[2][64 * 32];  // 2 x 4KB

  const int tid  = threadIdx.x;
  const int lane = tid & 31;
  const int wv   = tid >> 5;                 // wave in block: 0..3
  const int i0   = (blockIdx.x / tilesN64) * 64 + wv * 16;
  const int j0   = (blockIdx.x % tilesN64) * 64;
  const int half = lane >> 4;                // lanes 0-15 vs 16-31
  const int l15  = lane & 15;

  // --- B staging assignment: thread covers 32B = elems [sk, sk+16) of row sn
  const int sn = tid >> 1;                   // 0..63
  const int sk = (tid & 1) << 4;             // 0 or 16 (elements)
  const unsigned short* __restrict__ gsrc0 = Bt + (size_t)(j0 + sn) * ldb + sk;
  unsigned short* const lbuf[2] = {&Bs[0][sn * 32 + sk], &Bs[1][sn * 32 + sk]};

  auto stage = [&](int k0, int buf) {
    const unsigned short* g = gsrc0 + k0;
    unsigned short* l = lbuf[buf];
#if CA_HAVE_ASYNC_LDS
    // (global v4i* src, lds v4i* dst, imm offset [applies to both], imm cpol)
    __builtin_amdgcn_global_load_async_to_lds_b128(
        (__attribute__((address_space(1))) v4i*)g,
        (__attribute__((address_space(3))) v4i*)l, 0, 0);
    __builtin_amdgcn_global_load_async_to_lds_b128(
        (__attribute__((address_space(1))) v4i*)g,
        (__attribute__((address_space(3))) v4i*)l, 16, 0);
#else
    ((v4u*)l)[0] = ((const v4u*)g)[0];
    ((v4u*)l)[1] = ((const v4u*)g)[1];
#endif
  };

  const unsigned short* __restrict__ arow = A + (size_t)(i0 + l15) * lda;

  v8f acc[4];
#pragma unroll
  for (int t = 0; t < 4; ++t) acc[t] = (v8f){0.f, 0.f, 0.f, 0.f, 0.f, 0.f, 0.f, 0.f};

  const int nsteps = K >> 5;
  stage(0, 0);

  for (int s = 0; s < nsteps; ++s) {
    const int k0  = s << 5;
    const int cur = s & 1;

    if (s + 1 < nsteps) {
      stage(k0 + 32, cur ^ 1);               // prefetch next panel into LDS
#if CA_HAVE_ASYNC_LDS
      __builtin_amdgcn_s_wait_asynccnt(2);   // current panel landed; next in flight
#endif
    } else {
#if CA_HAVE_ASYNC_LDS
      __builtin_amdgcn_s_wait_asynccnt(0);
#endif
    }
    __syncthreads();                         // panel visible to all 4 waves

    // ---- A fragment: 16x32 bf16 (contiguous K = 16*half + 2v (+1)), direct global b128 ----
    v8u au;
#pragma unroll
    for (int v = 0; v < 8; ++v) {
      const int kk = 16 * half + 2 * v;
      au[v] = *(const unsigned int*)(arow + k0 + kk);
    }
    v16bf afrag = __builtin_bit_cast(v16bf, au);
    __builtin_prefetch(arow + k0 + 64, 0, 3);   // two K-steps ahead, near-cache

#pragma unroll
    for (int t = 0; t < 4; ++t) {
      // ---- B fragment from LDS: N = lane&15, K = 16*half + 2v (+1) ----
      const unsigned short* bp = &Bs[cur][(t * 16 + l15) * 32 + half * 16];
      v4u lo = ((const v4u*)bp)[0];
      v4u hi = ((const v4u*)bp)[1];
      v8u bu = {lo.x, lo.y, lo.z, lo.w, hi.x, hi.y, hi.z, hi.w};
      v16bf bfrag = __builtin_bit_cast(v16bf, bu);
      acc[t] = __builtin_amdgcn_wmma_f32_16x16x32_bf16(
          false, afrag, false, bfrag, (short)0, acc[t], false, false);
    }
    __syncthreads();                         // done reading before next overwrite
  }

  // ---- store C/D: VGPR v -> row (v + 8*half), lane -> column ----
#pragma unroll
  for (int t = 0; t < 4; ++t) {
    const int col = j0 + t * 16 + l15;
    const float cb = biasCol ? biasCol[col] : 0.f;
#pragma unroll
    for (int v = 0; v < 8; ++v) {
      const int row = i0 + v + 8 * half;
      const float rb = biasRow ? biasRow[row] : 0.f;
      C[(size_t)row * ldc + col] = acc[t][v] + rb + cb;
    }
  }
}

// ---------------------------------------------------------------------------
// Row softmax: P[row, :] = softmax(S[row, :]); output bf16. block per row.
// ---------------------------------------------------------------------------
__global__ __launch_bounds__(256)
void ca_softmax_row_kernel(const float* __restrict__ S,
                           unsigned short* __restrict__ P, int ncols) {
  __shared__ float red[256];
  const int row = blockIdx.x, tid = threadIdx.x;
  const float* __restrict__ srow = S + (size_t)row * ncols;

  float mx = -3.0e38f;
  for (int j = tid; j < ncols; j += 256) mx = fmaxf(mx, srow[j]);
  red[tid] = mx; __syncthreads();
  for (int off = 128; off > 0; off >>= 1) {
    if (tid < off) red[tid] = fmaxf(red[tid], red[tid + off]);
    __syncthreads();
  }
  mx = red[0]; __syncthreads();

  float sum = 0.f;
  for (int j = tid; j < ncols; j += 256) sum += __expf(srow[j] - mx);
  red[tid] = sum; __syncthreads();
  for (int off = 128; off > 0; off >>= 1) {
    if (tid < off) red[tid] += red[tid + off];
    __syncthreads();
  }
  const float inv = 1.f / red[0];
  for (int j = tid; j < ncols; j += 256)
    P[(size_t)row * ncols + j] = f2bf(__expf(srow[j] - mx) * inv);
}

// ---------------------------------------------------------------------------
// Column softmax: Pt[j, i] = softmax over i of S[i, j]; bf16 row-major output.
// ---------------------------------------------------------------------------
__global__ __launch_bounds__(256)
void ca_softmax_col_kernel(const float* __restrict__ S,
                           unsigned short* __restrict__ Pt,
                           int nrows, int ncols) {
  __shared__ float red[256];
  const int col = blockIdx.x, tid = threadIdx.x;

  float mx = -3.0e38f;
  for (int i = tid; i < nrows; i += 256) mx = fmaxf(mx, S[(size_t)i * ncols + col]);
  red[tid] = mx; __syncthreads();
  for (int off = 128; off > 0; off >>= 1) {
    if (tid < off) red[tid] = fmaxf(red[tid], red[tid + off]);
    __syncthreads();
  }
  mx = red[0]; __syncthreads();

  float sum = 0.f;
  for (int i = tid; i < nrows; i += 256) sum += __expf(S[(size_t)i * ncols + col] - mx);
  red[tid] = sum; __syncthreads();
  for (int off = 128; off > 0; off >>= 1) {
    if (tid < off) red[tid] += red[tid + off];
    __syncthreads();
  }
  const float inv = 1.f / red[0];
  for (int i = tid; i < nrows; i += 256)
    Pt[(size_t)col * nrows + i] = f2bf(__expf(S[(size_t)i * ncols + col] - mx) * inv);
}

// ---------------------------------------------------------------------------
// Final dot: sum over X_flat * ffn_w (deterministic two-stage reduction).
// X[r, :] = [u_p[r], a_alpha[r], u_p[r]*a_alpha[r], u_c[r]*a_beta[r]]
// ---------------------------------------------------------------------------
__global__ __launch_bounds__(256)
void ca_final_dot_kernel(const float* __restrict__ u_p, const float* __restrict__ u_c,
                         const float* __restrict__ aal, const float* __restrict__ abe,
                         const float* __restrict__ ffn_w,
                         float* __restrict__ partials) {
  __shared__ float red[256];
  const size_t total = (size_t)N_DIM * 4 * D_DIM;   // 8388608
  float s = 0.f;
  for (size_t e = (size_t)blockIdx.x * 256 + threadIdx.x; e < total;
       e += (size_t)gridDim.x * 256) {
    const int r   = (int)(e >> 13);     // / (4*D)
    const int c   = (int)(e & 8191);
    const int seg = c >> 11;            // / D
    const int cc  = c & 2047;
    const size_t idx = (size_t)r * D_DIM + cc;
    float x;
    if (seg == 0)      x = u_p[idx];
    else if (seg == 1) x = aal[idx];
    else if (seg == 2) x = u_p[idx] * aal[idx];
    else               x = u_c[idx] * abe[idx];
    s += x * ffn_w[e];
  }
  red[threadIdx.x] = s; __syncthreads();
  for (int off = 128; off > 0; off >>= 1) {
    if (threadIdx.x < off) red[threadIdx.x] += red[threadIdx.x + off];
    __syncthreads();
  }
  if (threadIdx.x == 0) partials[blockIdx.x] = red[0];
}

__global__ __launch_bounds__(256)
void ca_final_reduce_kernel(const float* __restrict__ partials,
                            const float* __restrict__ ffn_b,
                            float* __restrict__ out, int nparts) {
  __shared__ float red[256];
  float s = 0.f;
  for (int i = threadIdx.x; i < nparts; i += 256) s += partials[i];
  red[threadIdx.x] = s; __syncthreads();
  for (int off = 128; off > 0; off >>= 1) {
    if (threadIdx.x < off) red[threadIdx.x] += red[threadIdx.x + off];
    __syncthreads();
  }
  if (threadIdx.x == 0) {
    float v = red[0] + ffn_b[0];
    out[0] = v > 0.f ? v : 0.f;
  }
}

// ---------------------------------------------------------------------------
// Host side
// ---------------------------------------------------------------------------
extern "C" void kernel_launch(void* const* d_in, const int* in_sizes, int n_in,
                              void* d_out, int out_size, void* d_ws, size_t ws_size,
                              hipStream_t stream) {
  (void)in_sizes; (void)n_in; (void)out_size; (void)ws_size;
  const float* u_p   = (const float*)d_in[0];
  const float* u_c   = (const float*)d_in[1];
  const float* w_a   = (const float*)d_in[2];
  const float* ffn_w = (const float*)d_in[3];
  const float* ffn_b = (const float*)d_in[4];
  float* out = (float*)d_out;

  char* ws = (char*)d_ws;
  const size_t MB = 1u << 20;
  float*          S     = (float*)         (ws + 0 * MB);   // 4 MB (1024x1024 f32)
  unsigned short* Abf   = (unsigned short*)(ws + 4 * MB);   // 4 MB (1024x2048 bf16)
  unsigned short* Cbf   = (unsigned short*)(ws + 8 * MB);   // 4 MB (1024x2048 bf16)
  unsigned short* CbfT  = (unsigned short*)(ws + 12 * MB);  // 4 MB (2048x1024 bf16)
  unsigned short* PbfT  = (unsigned short*)(ws + 16 * MB);  // 4 MB (2048x1024 bf16)
  unsigned short* SalBf = (unsigned short*)(ws + 20 * MB);  // 2 MB (1024x1024 bf16)
  unsigned short* SbeBf = (unsigned short*)(ws + 22 * MB);  // 2 MB (1024x1024 bf16)
  float*          aal   = (float*)         (ws + 24 * MB);  // 8 MB (1024x2048 f32)
  float*          abe   = (float*)         (ws + 32 * MB);  // 8 MB (1024x2048 f32)
  float*          rp    = (float*)         (ws + 40 * MB);            // 4 KB
  float*          rc    = (float*)         (ws + 40 * MB + 4096);     // 4 KB
  float*          parts = (float*)         (ws + 40 * MB + 8192);     // 4 KB

  // 1) biases + bf16 operand staging
  ca_prep_kernel<<<dim3(1024, 2), 256, 0, stream>>>(u_p, u_c, w_a, Abf, Cbf,
                                                    CbfT, PbfT, rp, rc);

  // 2) S = rp[i] + rc[j] + (u_p*w3) @ u_c^T
  //    blocks = (1024/64)*(1024/64) = 256, tilesN64 = 16, K = 2048
  ca_gemm_bf16_kernel<<<256, 128, 0, stream>>>(Abf, D_DIM, Cbf, D_DIM,
                                               S, N_DIM, /*tilesN64=*/16,
                                               /*K=*/D_DIM, rp, rc);

  // 3) softmaxes of S (row-wise and column-wise) -> bf16 probability matrices
  ca_softmax_row_kernel<<<1024, 256, 0, stream>>>(S, SalBf, N_DIM);
  ca_softmax_col_kernel<<<1024, 256, 0, stream>>>(S, SbeBf, M_DIM, N_DIM);

  // 4) a_alpha = S_alpha @ u_c : blocks = (1024/64)*(2048/64) = 512
  ca_gemm_bf16_kernel<<<512, 128, 0, stream>>>(SalBf, N_DIM, CbfT, N_DIM,
                                               aal, D_DIM, /*tilesN64=*/32,
                                               /*K=*/N_DIM, nullptr, nullptr);

  // 5) a_beta = S_beta @ u_p
  ca_gemm_bf16_kernel<<<512, 128, 0, stream>>>(SbeBf, M_DIM, PbfT, M_DIM,
                                               abe, D_DIM, /*tilesN64=*/32,
                                               /*K=*/M_DIM, nullptr, nullptr);

  // 6) final dot with ffn_w, relu
  ca_final_dot_kernel<<<1024, 256, 0, stream>>>(u_p, u_c, aal, abe, ffn_w, parts);
  ca_final_reduce_kernel<<<1, 256, 0, stream>>>(parts, ffn_b, out, 1024);
}
